// MultiHeadSelfAttention_36919538877123
// MI455X (gfx1250) — matrix operand order repaired and checked by reading, not compiled, not run on hardware
//
#include <hip/hip_runtime.h>

#define DMODEL 1024
#define NHEADS 16
#define HD     64
#define BATCH  2
#define LSEQ   2048
#define MROWS  (BATCH * LSEQ)   // 4096
#define N_QKV  (3 * DMODEL)     // 3072
#define QB     2                // 16-row q-tiles per wave (paired: p and 127-p)

typedef __attribute__((ext_vector_type(16))) __bf16 v16bf;
typedef __attribute__((ext_vector_type(8)))  __bf16 v8bf;
typedef __attribute__((ext_vector_type(8)))  float  v8f;

static __device__ __forceinline__ v16bf cat16(v8bf lo, v8bf hi) {
  return __builtin_shufflevector(lo, hi, 0,1,2,3,4,5,6,7,8,9,10,11,12,13,14,15);
}

// A-operand fragment (16x32 bf16, contiguous k order matching B):
// lanes 0-15: K = kb+0..15; lanes 16-31: K = kb+16..31.
static __device__ __forceinline__ v16bf load_afrag(const __bf16* row, int kb, int hf) {
  v8bf lo = *(const v8bf*)(row + kb + hf * 16);
  v8bf hi = *(const v8bf*)(row + kb + hf * 16 + 8);
  return cat16(lo, hi);
}

// B-operand fragment (32x16 bf16): lane n holds column n; lanes 0-15: K 0-15,
// lanes 16-31: K 16-31. For C = A * B^T with B stored (N,K) row-major this is
// 16 contiguous bf16 from B-row n at column kb + hf*16.
static __device__ __forceinline__ v16bf load_bfrag(const __bf16* p) {
  v8bf lo = *(const v8bf*)p;
  v8bf hi = *(const v8bf*)(p + 8);
  return cat16(lo, hi);
}

static __device__ __forceinline__ v8f wmma_bf16(v16bf a, v16bf b, v8f c) {
  return __builtin_amdgcn_wmma_f32_16x16x32_bf16(false, a, false, b, (short)0, c,
                                                 false, false);
}

// ---------------------------------------------------------------- fp32 -> bf16
__global__ __launch_bounds__(256) void cvt_f32_bf16(const float* __restrict__ in,
                                                    __bf16* __restrict__ out, int n) {
  int i = blockIdx.x * blockDim.x + threadIdx.x;
  if (i < n) out[i] = (__bf16)in[i];
}

// -------------------------------------------- C[M,N] = A[M,K] * B[N,K]^T (bf16)
// One wave computes a 64x64 block (4x4 WMMA tiles): 262 KFLOP per 8 KB loaded.
template <typename OutT>
__global__ __launch_bounds__(256) void gemm_bf16_nt(const __bf16* __restrict__ A,
                                                    const __bf16* __restrict__ B,
                                                    OutT* __restrict__ C,
                                                    int N, int K) {
  const int lane = threadIdx.x & 31;
  const int wave = threadIdx.x >> 5;
  const int r  = lane & 15;
  const int hf = lane >> 4;
  const int n64  = N >> 6;
  const int unit = blockIdx.x * 8 + wave;   // over (M/64)*(N/64)
  const int tm = unit / n64;
  const int tn = unit % n64;

  const __bf16* arow0 = A + (size_t)(tm * 64 + r) * K;
  const __bf16* brow0 = B + (size_t)(tn * 64 + r) * K;
  const size_t step16 = (size_t)16 * K;

  v8f c[4][4];
#pragma unroll
  for (int s = 0; s < 4; ++s)
#pragma unroll
    for (int t = 0; t < 4; ++t) c[s][t] = (v8f){};

  for (int kb = 0; kb < K; kb += 32) {
    v16bf a[4], b[4];
    const int boff = kb + hf * 16;
#pragma unroll
    for (int s = 0; s < 4; ++s) a[s] = load_afrag(arow0 + (size_t)s * step16, kb, hf);
#pragma unroll
    for (int t = 0; t < 4; ++t) b[t] = load_bfrag(brow0 + (size_t)t * step16 + boff);
#pragma unroll
    for (int s = 0; s < 4; ++s)
#pragma unroll
      for (int t = 0; t < 4; ++t) c[s][t] = wmma_bf16(a[s], b[t], c[s][t]);
    __builtin_prefetch(arow0 + kb + 256, 0, 1);   // global_prefetch_b8
  }

  // C/D layout: lane holds rows m = v + hf*8, column n = r.
#pragma unroll
  for (int s = 0; s < 4; ++s) {
    OutT* crow = C + (size_t)(tm * 64 + s * 16 + hf * 8) * N + tn * 64 + r;
#pragma unroll
    for (int v = 0; v < 8; ++v) {
      OutT* p = crow + (size_t)v * N;
#pragma unroll
      for (int t = 0; t < 4; ++t) p[t * 16] = (OutT)c[s][t][v];
    }
  }
}

// --------------------------------------- RoPE + split heads + transpose V
// qkv: [MROWS, 3072] bf16.  Qr,Kr: [B*nh, L, 64].  Vt: [B*nh, 64, L].
// Q is pre-scaled by 1/sqrt(HD) so attention needs no per-score multiply.
__global__ __launch_bounds__(256) void rope_reshape(const __bf16* __restrict__ qkv,
                                                    __bf16* __restrict__ Qr,
                                                    __bf16* __restrict__ Kr,
                                                    __bf16* __restrict__ Vt) {
  int idx = blockIdx.x * blockDim.x + threadIdx.x;   // over MROWS*DMODEL
  if (idx >= MROWS * DMODEL) return;
  const int t = idx >> 10;
  const int c = idx & 1023;
  const int h = c >> 6;
  const int d = c & 63;
  const int b = t >> 11;
  const int l = t & 2047;
  const __bf16* row = qkv + (size_t)t * N_QKV;
  const float j = (float)(d & 31);
  const float inv_freq = __powf(10000.f, -j * (1.f / 32.f));
  const float theta = (float)l * inv_freq;
  const float cs = __cosf(theta), sn = __sinf(theta);
  const float q  = (float)row[c];
  const float k  = (float)row[DMODEL + c];
  const float vv = (float)row[2 * DMODEL + c];
  float qrot, krot;                         // rotate_half
  if (d < 32) { qrot = -(float)row[c + 32]; krot = -(float)row[DMODEL + c + 32]; }
  else        { qrot =  (float)row[c - 32]; krot =  (float)row[DMODEL + c - 32]; }
  const float qo = (q * cs + qrot * sn) * 0.125f;      // fold 1/sqrt(64)
  const float ko = k * cs + krot * sn;
  const size_t head = (size_t)(b * NHEADS + h);
  Qr[(head * LSEQ + l) * HD + d] = (__bf16)qo;
  Kr[(head * LSEQ + l) * HD + d] = (__bf16)ko;
  Vt[(head * HD + d) * LSEQ + l] = (__bf16)vv;
}

// ------------------------------------------------- causal flash attention
// One wave per (b, h, tile-pair). Each wave owns 16-row q-tiles p and 127-p,
// so every wave executes exactly 65 32-key steps (uniform makespan). K/V
// fragments are shared across both tiles while both are live; row-sums
// accumulate through a ones-column WMMA; causal masking only executes on
// diagonal-crossing steps.
__global__ __launch_bounds__(256) void flash_attn(const __bf16* __restrict__ Qr,
                                                  const __bf16* __restrict__ Kr,
                                                  const __bf16* __restrict__ Vt,
                                                  __bf16* __restrict__ Yb) {
  __shared__ __bf16 plds[8][QB][16 * 32];   // 16 KB: per-wave P staging tiles
  const int lane = threadIdx.x & 31;
  const int wave = threadIdx.x >> 5;
  const int r  = lane & 15;
  const int hf = lane >> 4;
  const int unit = blockIdx.x * 8 + wave;   // 2*16*64 = 2048 units
  const int b   = unit >> 10;
  const int rem = unit & 1023;
  const int h  = rem >> 6;
  const int pb = rem & 63;                  // tile-pair index
  const int rowb[QB] = { pb * 16, (127 - pb) * 16 };   // paired q-tile bases
  const int njt0 = (pb >> 1) + 1;                      // steps tile0 needs
  const int njt  = ((127 - pb) >> 1) + 1;              // steps tile1 needs
  const size_t head = (size_t)(b * NHEADS + h);
  const __bf16* Qh = Qr + head * LSEQ * HD;
  const __bf16* Kh = Kr + head * LSEQ * HD;
  const __bf16* Vh = Vt + head * HD * LSEQ;

  v16bf aq0[QB], aq1[QB];
#pragma unroll
  for (int qi = 0; qi < QB; ++qi) {
    const int qrow = rowb[qi] + r;
    aq0[qi] = load_afrag(Qh + (size_t)qrow * HD, 0, hf);
    aq1[qi] = load_afrag(Qh + (size_t)qrow * HD, 32, hf);
  }

  v16bf ones;
#pragma unroll
  for (int i = 0; i < 16; ++i) ones[i] = (__bf16)1.0f;

  float mstat[QB][8];
  v8f o[QB][4], ol[QB];
#pragma unroll
  for (int qi = 0; qi < QB; ++qi) {
#pragma unroll
    for (int v = 0; v < 8; ++v) mstat[qi][v] = -1e30f;
#pragma unroll
    for (int t = 0; t < 4; ++t) o[qi][t] = (v8f){};
    ol[qi] = (v8f){};
  }

  for (int j = 0; j < njt; ++j) {
    const int kbase = j * 32;
    // Shared K fragments: 2 key n-tiles x 2 k-steps
    const __bf16* k0r = Kh + (size_t)(kbase + r) * HD + hf * 16;
    const __bf16* k1r = Kh + (size_t)(kbase + 16 + r) * HD + hf * 16;
    const v16bf bk00 = load_bfrag(k0r);
    const v16bf bk01 = load_bfrag(k0r + 32);
    const v16bf bk10 = load_bfrag(k1r);
    const v16bf bk11 = load_bfrag(k1r + 32);

#pragma unroll
    for (int qi = 0; qi < QB; ++qi) {
      if (qi == 0 && j >= njt0) continue;   // tile0 already complete (uniform)
      v8f s0 = (v8f){}, s1 = (v8f){};
      s0 = wmma_bf16(aq0[qi], bk00, s0);
      s0 = wmma_bf16(aq1[qi], bk01, s0);
      s1 = wmma_bf16(aq0[qi], bk10, s1);
      s1 = wmma_bf16(aq1[qi], bk11, s1);

      const int qmin = rowb[qi];
      if (kbase + 31 > qmin) {              // diagonal-crossing step: mask
#pragma unroll
        for (int v = 0; v < 8; ++v) {
          const int q = qmin + v + hf * 8;
          s0[v] = (kbase + r      > q) ? -1e30f : s0[v];
          s1[v] = (kbase + 16 + r > q) ? -1e30f : s1[v];
        }
      }

      __bf16* pw = &plds[wave][qi][0];
      float alpha[8];
#pragma unroll
      for (int v = 0; v < 8; ++v) {
        float tmax = fmaxf(s0[v], s1[v]);
#pragma unroll
        for (int mk = 8; mk >= 1; mk >>= 1)
          tmax = fmaxf(tmax, __shfl_xor(tmax, mk, 32));  // row spans 16 lanes
        const float nm = fmaxf(mstat[qi][v], tmax);
        alpha[v] = __expf(mstat[qi][v] - nm);
        const float p0 = __expf(s0[v] - nm);
        const float p1 = __expf(s1[v] - nm);
        mstat[qi][v] = nm;
        const int m = v + hf * 8;                        // stage P (16x32)
        pw[m * 32 + r]      = (__bf16)p0;
        pw[m * 32 + 16 + r] = (__bf16)p1;
      }
#pragma unroll
      for (int v = 0; v < 8; ++v) {
#pragma unroll
        for (int t = 0; t < 4; ++t) o[qi][t][v] *= alpha[v];
        ol[qi][v] *= alpha[v];
      }
    }

    // same-wave LDS RAW: drain DS counter before re-reading in A-layout
    asm volatile("s_wait_dscnt 0" ::: "memory");

    // Shared V fragments
    const __bf16* vb = Vh + (size_t)r * LSEQ + kbase + hf * 16;
    v16bf bv[4];
#pragma unroll
    for (int t = 0; t < 4; ++t) bv[t] = load_bfrag(vb + (size_t)(t * 16) * LSEQ);

#pragma unroll
    for (int qi = 0; qi < QB; ++qi) {
      if (qi == 0 && j >= njt0) continue;   // tile0 already complete (uniform)
      const __bf16* pw = &plds[wave][qi][0];
      v8bf plo = *(const v8bf*)&pw[r * 32 + hf * 16];
      v8bf phi = *(const v8bf*)&pw[r * 32 + hf * 16 + 8];
      const v16bf ap = cat16(plo, phi);
#pragma unroll
      for (int t = 0; t < 4; ++t) o[qi][t] = wmma_bf16(ap, bv[t], o[qi][t]);
      ol[qi] = wmma_bf16(ap, ones, ol[qi]);  // row-sum accumulator
    }
  }

#pragma unroll
  for (int qi = 0; qi < QB; ++qi) {
    __bf16* yrow = Yb + (size_t)(b * LSEQ + rowb[qi] + hf * 8) * DMODEL
                 + h * HD + r;
#pragma unroll
    for (int v = 0; v < 8; ++v) {
      const float inv = 1.f / ol[qi][v];
      __bf16* p = yrow + (size_t)v * DMODEL;
#pragma unroll
      for (int t = 0; t < 4; ++t) p[t * 16] = (__bf16)(o[qi][t][v] * inv);
    }
  }
}

// --------------------------------------------------------------------- launch
extern "C" void kernel_launch(void* const* d_in, const int* in_sizes, int n_in,
                              void* d_out, int out_size, void* d_ws, size_t ws_size,
                              hipStream_t stream) {
  (void)in_sizes; (void)n_in; (void)out_size; (void)ws_size;
  const float* x    = (const float*)d_in[0];   // (2, 2048, 1024)
  const float* wqkv = (const float*)d_in[1];   // (3072, 1024)
  const float* wo   = (const float*)d_in[2];   // (1024, 1024)
  float* out = (float*)d_out;                  // (2, 2048, 1024) fp32

  char* ws = (char*)d_ws;                      // all offsets 16B+ aligned
  __bf16* xb    = (__bf16*)(ws + 0);                       //  8,388,608 B
  __bf16* wqkvb = (__bf16*)(ws + 8388608);                 //  6,291,456 B
  __bf16* wob   = (__bf16*)(ws + 14680064);                //  2,097,152 B
  __bf16* qkvb  = (__bf16*)(ws + 16777216);                // 25,165,824 B
  __bf16* Qr    = (__bf16*)(ws + 41943040);                //  8,388,608 B
  __bf16* Kr    = (__bf16*)(ws + 50331648);                //  8,388,608 B
  __bf16* Vt    = (__bf16*)(ws + 58720256);                //  8,388,608 B
  __bf16* Yb    = (__bf16*)(ws + 67108864);                //  8,388,608 B

  const int nx = MROWS * DMODEL;          // 4,194,304
  const int nw = N_QKV * DMODEL;          // 3,145,728
  const int no = DMODEL * DMODEL;         // 1,048,576
  cvt_f32_bf16<<<(nx + 255) / 256, 256, 0, stream>>>(x, xb, nx);
  cvt_f32_bf16<<<(nw + 255) / 256, 256, 0, stream>>>(wqkv, wqkvb, nw);
  cvt_f32_bf16<<<(no + 255) / 256, 256, 0, stream>>>(wo, wob, no);

  // qkv = x @ wqkv^T : (4096/64)*(3072/64) = 3072 wave-units -> 384 blocks
  gemm_bf16_nt<__bf16><<<384, 256, 0, stream>>>(xb, wqkvb, qkvb, N_QKV, DMODEL);

  rope_reshape<<<nx / 256, 256, 0, stream>>>(qkvb, Qr, Kr, Vt);

  // 2 * 16 * 64 tile-pairs = 2048 wave-units -> 256 blocks, uniform 65 steps
  flash_attn<<<256, 256, 0, stream>>>(Qr, Kr, Vt, Yb);

  // out = Y @ wo^T : (4096/64)*(1024/64) = 1024 wave-units -> 128 blocks
  gemm_bf16_nt<float><<<128, 256, 0, stream>>>(Yb, wob, out, DMODEL, DMODEL);
}
